// DifferentialEntropyRegularization_16716012716655
// MI455X (gfx1250) — matrix operand order repaired and checked by reading, not compiled, not run on hardware
//
#include <hip/hip_runtime.h>
#include <hip/hip_bf16.h>

// ---------------------------------------------------------------------------
// Problem constants (reference: x is [16384, 512] fp32)
// ---------------------------------------------------------------------------
#define NROWS 16384
#define DDIM  512
#define ROWS_PER_WG 64      // 4 waves * 16 rows
#define WG_THREADS 128      // 4 wave32
#define COL_TILE 16
#define NTILES (NROWS / COL_TILE)

// Converted layout: per row, 2048 bytes = [ hi bf16 x512 | lo bf16 x512 ]
#define ROW_USHORTS (2 * DDIM)          // 1024
#define A_LDS_USHORTS (ROWS_PER_WG * ROW_USHORTS)   // 65536 (128KB)
#define B_LDS_USHORTS (COL_TILE * ROW_USHORTS)      // 16384 (32KB)

typedef __attribute__((ext_vector_type(16))) __bf16 v16bf;
typedef __attribute__((ext_vector_type(8)))  float  v8f;
typedef __attribute__((ext_vector_type(4)))  unsigned int u32x4;
typedef __attribute__((ext_vector_type(8)))  int i32x8;
typedef __attribute__((ext_vector_type(4)))  int i32x4;

union FragBF16 {
    uint4 u[2];   // 32 bytes
    v16bf v;
};

// float -> bf16 round-to-nearest-even (data is finite normals; no NaN path)
__device__ __forceinline__ unsigned short f2bf(float f) {
    unsigned int u = __float_as_uint(f);
    unsigned int r = u + 0x7FFFu + ((u >> 16) & 1u);
    return (unsigned short)(r >> 16);
}
__device__ __forceinline__ float bf2f(unsigned short h) {
    return __uint_as_float(((unsigned int)h) << 16);
}

#if defined(__has_builtin)
#if __has_builtin(__builtin_amdgcn_tensor_load_to_lds) && __has_builtin(__builtin_amdgcn_s_wait_tensorcnt)
#define HAVE_TDM 1
#endif
#endif

#ifdef HAVE_TDM
// Issue one TDM descriptor copying a contiguous region (n8 x 8 bytes) from
// global memory to LDS byte offset ldsOff. 2D degenerate tile: 1 row of n8
// 8-byte elements. Must be issued with the whole wave converged.
__device__ __forceinline__ void tdm_load_contig(unsigned ldsOff,
                                                const void* gsrc,
                                                unsigned n8) {
    unsigned long long ga = (unsigned long long)(size_t)gsrc;
    u32x4 g0;
    g0[0] = 1u;                                          // count=1, no gather
    g0[1] = ldsOff;                                      // lds_addr (bytes)
    g0[2] = (unsigned)(ga & 0xFFFFFFFFull);              // global_addr[31:0]
    g0[3] = (unsigned)((ga >> 32) & 0x01FFFFFFull)       // global_addr[56:32]
            | (2u << 30);                                // type = 2 ("image")
    i32x8 g1;
    g1[0] = (3 << 16);                                   // data_size=3 (8B), mask=0
    g1[1] = (int)((n8 & 0xFFFFu) << 16);                 // tensor_dim0[15:0]
    g1[2] = (int)(((n8 >> 16) & 0xFFFFu) | (1u << 16));  // tensor_dim0[31:16], tensor_dim1=1
    g1[3] = (int)((n8 & 0xFFFFu) << 16);                 // tile_dim0 = n8
    g1[4] = 1;                                           // tile_dim1 = 1
    g1[5] = (int)n8;                                     // tensor_dim0_stride[31:0]
    g1[6] = 0;                                           // stride hi / dim1_stride lo
    g1[7] = 0;
    i32x4 gz = {0, 0, 0, 0};
#if __clang_major__ >= 23
    i32x8 gz8 = {0, 0, 0, 0, 0, 0, 0, 0};
    __builtin_amdgcn_tensor_load_to_lds(g0, g1, gz, gz, gz8, 0);
#else
    __builtin_amdgcn_tensor_load_to_lds(g0, g1, gz, gz, 0);
#endif
}
#endif

// ---------------------------------------------------------------------------
// Kernel 0: split fp32 -> (hi, lo) bf16, interleaved per row:
//   xcvt[row] = [ hi[0..511] | lo[0..511] ]  (2048 bytes per row)
// ---------------------------------------------------------------------------
__global__ void convert_kernel(const float* __restrict__ x,
                               unsigned short* __restrict__ xcvt) {
    const int idx = blockIdx.x * blockDim.x + threadIdx.x; // float4 group id
    const int r  = idx >> (9 - 2);                         // / (DDIM/4)
    const int k4 = idx & (DDIM / 4 - 1);
    float4 v = ((const float4*)x)[idx];
    ushort4 hi, lo;
    hi.x = f2bf(v.x); lo.x = f2bf(v.x - bf2f(hi.x));
    hi.y = f2bf(v.y); lo.y = f2bf(v.y - bf2f(hi.y));
    hi.z = f2bf(v.z); lo.z = f2bf(v.z - bf2f(hi.z));
    hi.w = f2bf(v.w); lo.w = f2bf(v.w - bf2f(hi.w));
    ushort4* rowp = (ushort4*)(xcvt + (size_t)r * ROW_USHORTS);
    rowp[k4] = hi;
    rowp[DDIM / 4 + k4] = lo;
}

// ---------------------------------------------------------------------------
// Kernel 1: fused x @ x^T (split bf16 WMMA) + masked row argmax.
//   grid:  NROWS/ROWS_PER_WG = 256 blocks, 128 threads (4 waves)
//   LDS:   A block 128KB + double-buffered B tile 2x32KB = 192KB (dynamic)
//   Staging via TDM async copies, overlapped with WMMA compute.
// ---------------------------------------------------------------------------
__global__ void nn_argmax_kernel(const unsigned short* __restrict__ xcvt,
                                 int* __restrict__ outIdx) {
    extern __shared__ unsigned short smem[];
    unsigned short* Abuf = smem;                       // [64][1024]
    unsigned short* Bbuf = smem + A_LDS_USHORTS;       // 2 x [16][1024]

    const int tid  = threadIdx.x;
    const int wave = tid >> 5;
    const int lane = tid & 31;
    const int rowBase = blockIdx.x * ROWS_PER_WG;

    const int m    = lane & 15;
    const int half = lane >> 4;

#ifdef HAVE_TDM
    // ---- async stage: A block (128KB) + first B tile (32KB) ----
    if (wave == 0) {
        tdm_load_contig(0u, xcvt + (size_t)rowBase * ROW_USHORTS,
                        A_LDS_USHORTS / 4);
        tdm_load_contig((unsigned)(A_LDS_USHORTS * 2),
                        xcvt, B_LDS_USHORTS / 4);
        __builtin_amdgcn_s_wait_tensorcnt(0);
    }
    __syncthreads();
#else
    for (int i = tid; i < A_LDS_USHORTS; i += WG_THREADS)
        Abuf[i] = xcvt[(size_t)rowBase * ROW_USHORTS + i];
    for (int i = tid; i < B_LDS_USHORTS; i += WG_THREADS)
        Bbuf[i] = xcvt[i];
    __syncthreads();
#endif

    float bestV[8];
    int   bestI[8];
#pragma unroll
    for (int r = 0; r < 8; ++r) { bestV[r] = -3.0e38f; bestI[r] = 0; }

    const unsigned short* aRowHi = Abuf + (size_t)(wave * 16 + m) * ROW_USHORTS;
    const unsigned short* aRowLo = aRowHi + DDIM;

    for (int jt = 0; jt < NTILES; ++jt) {
        const int cur = jt & 1;
        const int colBase = jt * COL_TILE;

#ifdef HAVE_TDM
        // prefetch next B tile into the other buffer while we compute
        if (wave == 0 && (jt + 1) < NTILES) {
            tdm_load_contig((unsigned)((A_LDS_USHORTS + (1 - cur) * B_LDS_USHORTS) * 2),
                            xcvt + (size_t)(jt + 1) * B_LDS_USHORTS,
                            B_LDS_USHORTS / 4);
        }
#endif

        const unsigned short* bRowHi = Bbuf + cur * B_LDS_USHORTS
                                     + (size_t)m * ROW_USHORTS;
        const unsigned short* bRowLo = bRowHi + DDIM;

        v8f acc = {};
        // ---- K loop: 16 chunks of K=32, 3 bf16 WMMAs each (hh + hl + lh) ----
        for (int kc = 0; kc < DDIM; kc += 32) {
            FragBF16 ahi, alo, bhi, blo;
            // A 16x32 bf16 layout: lane half selects K+8*half (v0..3) and
            // K+16+8*half (v4..7), pairs of consecutive K per VGPR.
            ahi.u[0] = *(const uint4*)(aRowHi + kc + 8 * half);
            ahi.u[1] = *(const uint4*)(aRowHi + kc + 16 + 8 * half);
            alo.u[0] = *(const uint4*)(aRowLo + kc + 8 * half);
            alo.u[1] = *(const uint4*)(aRowLo + kc + 16 + 8 * half);
            // B 32x16 bf16 layout: same K order as A (K+8*half, K+16+8*half).
            bhi.u[0] = *(const uint4*)(bRowHi + kc + 8 * half);
            bhi.u[1] = *(const uint4*)(bRowHi + kc + 8 * half + 16);
            blo.u[0] = *(const uint4*)(bRowLo + kc + 8 * half);
            blo.u[1] = *(const uint4*)(bRowLo + kc + 8 * half + 16);

            acc = __builtin_amdgcn_wmma_f32_16x16x32_bf16(
                false, ahi.v, false, bhi.v, (short)0, acc, false, false);
            acc = __builtin_amdgcn_wmma_f32_16x16x32_bf16(
                false, ahi.v, false, blo.v, (short)0, acc, false, false);
            acc = __builtin_amdgcn_wmma_f32_16x16x32_bf16(
                false, alo.v, false, bhi.v, (short)0, acc, false, false);
        }

        // ---- epilogue: masked running argmax ----
        const int col  = colBase + m;
        const int rowB = rowBase + wave * 16 + 8 * half;
#pragma unroll
        for (int r = 0; r < 8; ++r) {
            const int row = rowB + r;
            float v = (row == col) ? -1.0f : acc[r];   // diagonal mask = -1.0
            if (v > bestV[r]) { bestV[r] = v; bestI[r] = col; }
        }

#ifdef HAVE_TDM
        if (wave == 0) __builtin_amdgcn_s_wait_tensorcnt(0);
        __syncthreads();   // prefetched tile visible; all done with 'cur'
#else
        __syncthreads();
        for (int i = tid; i < B_LDS_USHORTS; i += WG_THREADS) {
            int nj = (jt + 1 < NTILES) ? (jt + 1) : jt;
            Bbuf[((jt + 1) & 1) * B_LDS_USHORTS + i] =
                xcvt[(size_t)nj * B_LDS_USHORTS + i];
        }
        __syncthreads();
#endif
    }

    // ---- cross-lane argmax over the 16 lanes sharing each row ----
#pragma unroll
    for (int r = 0; r < 8; ++r) {
        float v = bestV[r];
        int   idx = bestI[r];
#pragma unroll
        for (int mm = 8; mm >= 1; mm >>= 1) {
            float ov = __shfl_xor(v, mm, 32);
            int   oi = __shfl_xor(idx, mm, 32);
            if (ov > v || (ov == v && oi < idx)) { v = ov; idx = oi; }
        }
        if (m == 0) {
            const int row = rowBase + wave * 16 + 8 * half + r;
            outIdx[row] = idx;
        }
    }
}

// ---------------------------------------------------------------------------
// Kernel 2: one wave per row -> log(||x[r] - x[I[r]] + 1e-6||_2 + 1e-8)
// ---------------------------------------------------------------------------
__global__ void row_logrho_kernel(const float* __restrict__ x,
                                  const int* __restrict__ idx,
                                  float* __restrict__ t) {
    const int wave = threadIdx.x >> 5;
    const int lane = threadIdx.x & 31;
    const int row  = blockIdx.x * 8 + wave;
    if (row >= NROWS) return;

    const float* xr = x + (size_t)row * DDIM;
    const float* xn = x + (size_t)idx[row] * DDIM;
    float s = 0.0f;
    for (int k = lane; k < DDIM; k += 32) {
        float d = xr[k] - xn[k] + 1.0e-6f;
        s += d * d;
    }
#pragma unroll
    for (int mm = 16; mm >= 1; mm >>= 1) s += __shfl_xor(s, mm, 32);
    if (lane == 0) {
        float rho = sqrtf(s);
        t[row] = logf(rho + 1.0e-8f);
    }
}

// ---------------------------------------------------------------------------
// Kernel 3: deterministic single-block reduction: out = -mean(t)
// ---------------------------------------------------------------------------
__global__ void reduce_loss_kernel(const float* __restrict__ t,
                                   float* __restrict__ out) {
    __shared__ float sh[256];
    float s = 0.0f;
    for (int i = threadIdx.x; i < NROWS; i += 256) s += t[i];
    sh[threadIdx.x] = s;
    __syncthreads();
    for (int st = 128; st >= 1; st >>= 1) {
        if (threadIdx.x < st) sh[threadIdx.x] += sh[threadIdx.x + st];
        __syncthreads();
    }
    if (threadIdx.x == 0) out[0] = -sh[0] / (float)NROWS;
}

// ---------------------------------------------------------------------------
extern "C" void kernel_launch(void* const* d_in, const int* in_sizes, int n_in,
                              void* d_out, int out_size, void* d_ws, size_t ws_size,
                              hipStream_t stream) {
    (void)in_sizes; (void)n_in; (void)out_size; (void)ws_size;

    const float* x = (const float*)d_in[0];
    float* out = (float*)d_out;

    // Workspace layout: xcvt (32MB) | I (64KB) | T (64KB)
    unsigned short* xcvt = (unsigned short*)d_ws;
    const size_t xcvtBytes = (size_t)NROWS * ROW_USHORTS * sizeof(unsigned short);
    int*   I = (int*)((char*)d_ws + xcvtBytes);
    float* T = (float*)((char*)d_ws + xcvtBytes + NROWS * sizeof(int));

    const size_t smemBytes =
        (size_t)(A_LDS_USHORTS + 2 * B_LDS_USHORTS) * sizeof(unsigned short); // 192 KB

    static bool attrSet = false;  // host-side, deterministic, capture-safe
    if (!attrSet) {
        hipFuncSetAttribute((const void*)nn_argmax_kernel,
                            hipFuncAttributeMaxDynamicSharedMemorySize,
                            (int)smemBytes);
        attrSet = true;
    }

    convert_kernel<<<(NROWS * DDIM / 4) / 256, 256, 0, stream>>>(x, xcvt);
    nn_argmax_kernel<<<NROWS / ROWS_PER_WG, WG_THREADS, smemBytes, stream>>>(xcvt, I);
    row_logrho_kernel<<<NROWS / 8, 256, 0, stream>>>(x, I, T);
    reduce_loss_kernel<<<1, 256, 0, stream>>>(T, out);
}
